// Block_11020886082299
// MI455X (gfx1250) — 8-bit matrix operand order repaired and checked by reading, compiled, not run on hardware
//
#include <hip/hip_runtime.h>

typedef __attribute__((ext_vector_type(8))) int v8i;

#define DIM_IN  4096
#define DIM_OUT 4096
#define BATCH   64
#define KSPLIT  4
#define KCHUNK  64
#define KPER    (DIM_IN / KSPLIT)
#define NCHUNKS (KPER / KCHUNK)
#define BT_STRIDE 80
#define BT_TILE   (16 * BT_STRIDE)

__global__ void __launch_bounds__(256)
pack_x_kernel(const int* __restrict__ x,
              unsigned char* __restrict__ xpk,
              int* __restrict__ sumx) {
    const int b = blockIdx.x;
    const int t = threadIdx.x;
    const int* xr = x + (size_t)b * DIM_IN;
    const int base = t * 16;

    int s = 0;
    unsigned w[4];
#pragma unroll
    for (int g = 0; g < 4; ++g) {
        int4 v = *(const int4*)(xr + base + g * 4);
        unsigned b0 = (unsigned)(v.x & 1);
        unsigned b1 = (unsigned)(v.y & 1);
        unsigned b2 = (unsigned)(v.z & 1);
        unsigned b3 = (unsigned)(v.w & 1);
        w[g] = b0 | (b1 << 8) | (b2 << 16) | (b3 << 24);
        s += (int)(b0 + b1 + b2 + b3);
    }
    *(uint4*)(xpk + (size_t)b * DIM_IN + base) = make_uint4(w[0], w[1], w[2], w[3]);

    __shared__ int red[256];
    red[t] = s;
    __syncthreads();
    for (int off = 128; off > 0; off >>= 1) {
        if (t < off) red[t] += red[t + off];
        __syncthreads();
    }
    if (t == 0) sumx[b] = red[0];
}

__global__ void __launch_bounds__(512)
bnn_wmma_kernel(const unsigned char* __restrict__ xpk,
                const int* __restrict__ masks,
                const int* __restrict__ thr,
                const int* __restrict__ sumx,
                float* __restrict__ out) {
    __shared__ __align__(16) unsigned char Bt[KSPLIT * 2 * BT_TILE];
    __shared__ int redacc[(KSPLIT - 1) * 4 * 256];
    __shared__ int colpart[KSPLIT * 16];

    const int t    = threadIdx.x;
    const int lane = t & 31;
    const int wave = t >> 5;
    const int mt   = wave & 3;
    const int ks   = wave >> 2;
    const int half = lane >> 4;
    const int nl   = lane & 15;
    const int n0   = blockIdx.x * 16;
    const int k0s  = ks * KPER;

    const int ts = t & 127;
    const int ns = ts & 15;
    const int kk = ts >> 4;

    const int* mbase =
        masks + (size_t)(k0s + kk * 8) * DIM_OUT + n0 + ns;
    unsigned char* btw = Bt + ks * 2 * BT_TILE + ns * BT_STRIDE + kk * 8;
    const unsigned char* btr = Bt + ks * 2 * BT_TILE + nl * BT_STRIDE;
    const unsigned char* arow =
        xpk + (size_t)(mt * 16 + nl) * DIM_IN + k0s + half * 16;

    v8i acc  = (v8i)0;
    v8i acc1 = (v8i)0;
    const v8i aone = (v8i)(0x01010101);

    auto stage = [&](int chunk, int buf) {
        const int* mp = mbase + (size_t)chunk * KCHUNK * DIM_OUT;
        unsigned w0 = (unsigned)(mp[0] & 1)
                    | ((unsigned)(mp[1 * DIM_OUT] & 1) << 8)
                    | ((unsigned)(mp[2 * DIM_OUT] & 1) << 16)
                    | ((unsigned)(mp[3 * DIM_OUT] & 1) << 24);
        unsigned w1 = (unsigned)(mp[4 * DIM_OUT] & 1)
                    | ((unsigned)(mp[5 * DIM_OUT] & 1) << 8)
                    | ((unsigned)(mp[6 * DIM_OUT] & 1) << 16)
                    | ((unsigned)(mp[7 * DIM_OUT] & 1) << 24);
        *(uint2*)(btw + buf * BT_TILE) = make_uint2(w0, w1);
    };

    stage(0, 0);

    for (int i = 0; i < NCHUNKS; ++i) {
        const int buf = i & 1;
        __syncthreads();
        if (i + 1 < NCHUNKS) stage(i + 1, buf ^ 1);

        const unsigned char* bp = btr + buf * BT_TILE + half * 16;
        uint4 lo = *(const uint4*)(bp);
        uint4 hi = *(const uint4*)(bp + 32);
        v8i bmat = (v8i){(int)lo.x, (int)lo.y, (int)lo.z, (int)lo.w,
                         (int)hi.x, (int)hi.y, (int)hi.z, (int)hi.w};

        const unsigned char* ap = arow + i * KCHUNK;
        uint2 a01 = *(const uint2*)(ap);
        uint2 a23 = *(const uint2*)(ap + 8);
        uint2 a45 = *(const uint2*)(ap + 32);
        uint2 a67 = *(const uint2*)(ap + 40);
        v8i amat = (v8i){(int)a01.x, (int)a01.y, (int)a23.x, (int)a23.y,
                         (int)a45.x, (int)a45.y, (int)a67.x, (int)a67.y};

        acc = __builtin_amdgcn_wmma_i32_16x16x64_iu8(
                  false, amat, false, bmat, acc, false, false);
        acc1 = __builtin_amdgcn_wmma_i32_16x16x64_iu8(
                  false, aone, false, bmat, acc1, false, false);
    }

    if (ks > 0) {
        int* dst = redacc + ((ks - 1) * 4 + mt) * 256 + (half * 8) * 16 + nl;
#pragma unroll
        for (int r = 0; r < 8; ++r) dst[r * 16] = acc[r];
    }
    if (mt == 0 && half == 0) colpart[ks * 16 + nl] = acc1[0];
    __syncthreads();

    if (ks == 0) {
        const int cs = colpart[0 * 16 + nl] + colpart[1 * 16 + nl] +
                       colpart[2 * 16 + nl] + colpart[3 * 16 + nl];
        const int tn = thr[n0 + nl];
#pragma unroll
        for (int r = 0; r < 8; ++r) {
            int v = acc[r];
            const int off = mt * 256 + (half * 8 + r) * 16 + nl;
            v += redacc[0 * 1024 + off] + redacc[1 * 1024 + off] +
                 redacc[2 * 1024 + off];
            const int b = mt * 16 + half * 8 + r;
            const int s = DIM_IN - sumx[b] - cs + 2 * v;
            out[(size_t)b * DIM_OUT + n0 + nl] = (s > tn) ? 1.0f : 0.0f;
        }
    }
}

extern "C" void kernel_launch(void* const* d_in, const int* in_sizes, int n_in,
                              void* d_out, int out_size, void* d_ws, size_t ws_size,
                              hipStream_t stream) {
    const int* x     = (const int*)d_in[0];
    const int* masks = (const int*)d_in[1];
    const int* thr   = (const int*)d_in[2];
    float* out       = (float*)d_out;

    unsigned char* xpk = (unsigned char*)d_ws;
    int* sumx          = (int*)((char*)d_ws + BATCH * DIM_IN);

    pack_x_kernel<<<BATCH, 256, 0, stream>>>(x, xpk, sumx);
    bnn_wmma_kernel<<<DIM_OUT / 16, 512, 0, stream>>>(xpk, masks, thr, sumx, out);
}
